// Transformer_29738353557803
// MI455X (gfx1250) — matrix operand order repaired and checked by reading, not compiled, not run on hardware
//
#include <hip/hip_runtime.h>
#include <hip/hip_bf16.h>
#include <math.h>

// Problem constants (from reference)
#define S_ 1024
#define D_ 1024
#define H_ 16
#define HD_ 64
#define L_ 64
#define E_ 8
#define F_ 4096

typedef __bf16 bf16x16 __attribute__((ext_vector_type(16)));
typedef float  f32x8   __attribute__((ext_vector_type(8)));
typedef unsigned short u16;

#if defined(__gfx1250__) && __has_builtin(__builtin_amdgcn_global_load_async_to_lds_b128)
#define USE_ASYNC_LDS 1
typedef int int4v __attribute__((vector_size(16)));
typedef int4v __attribute__((address_space(1)))* gvec_t;   // global src (b128)
typedef int4v __attribute__((address_space(3)))* lvec_t;   // LDS dst (b128)
#else
#define USE_ASYNC_LDS 0
#endif

__device__ __forceinline__ void wait_async0() {
#if defined(__gfx1250__)
#if __has_builtin(__builtin_amdgcn_s_wait_asynccnt)
  __builtin_amdgcn_s_wait_asynccnt(0);
#else
  asm volatile("s_wait_asynccnt 0x0" ::: "memory");
#endif
#endif
}

__device__ __forceinline__ u16 f2bf(float f) {
  unsigned int u = __float_as_uint(f);
  u += 0x7fffu + ((u >> 16) & 1u);
  return (u16)(u >> 16);
}

union FragAB { uint4 u[2]; bf16x16 v; };
union Pack16 { uint4 q[2]; u16 h[16]; };

// ---------------------------------------------------------------------------
// Generic batched bf16 WMMA GEMM (REQUIRES: K%32==0, M%16==0, N%16==0):
//   C[M,N] (f32, optional) = A[M,K] x op(B) (+ C if accumulate) (+ bias) (gelu)
//   op(B) = transB ? B[N,K] : B[K,N]     (all bf16 row-major)
// Block tile 128x128, K-step 32, 8 wave32 waves (4 M-slices x 2 N-slices),
// 2x4 v_wmma_f32_16x16x32_bf16 accumulators per wave.
// Double-buffered LDS pipeline; global->LDS staging is ASYNCcnt-tracked
// (global_load_async_to_lds_b128) when the toolchain exposes the builtin.
// ---------------------------------------------------------------------------
__global__ __launch_bounds__(256) void wmma_gemm(
    const u16* __restrict__ A, long long aBatch, int lda,
    const u16* __restrict__ B, long long bBatch, int ldb, int transB,
    float* __restrict__ C, long long cBatch, int ldc,
    u16* __restrict__ Cbf, long long cbfBatch, int ldcbf,
    const float* __restrict__ bias, long long biasBatch,
    int M, int N, int Kd, int accumulate, int act)
{
  __shared__ u16 sA[2][128 * 32];   // [buf][row][k]
  __shared__ u16 sB[2][128 * 32];   // [buf][col][k]   (n-major, K contiguous)

  const int tid  = threadIdx.x;
  const int lane = tid & 31;
  const int wave = tid >> 5;
  const int wm = wave >> 1;         // 0..3 : 32-row slice of block tile
  const int wn = wave & 1;          // 0..1 : 64-col slice of block tile
  const int lr = lane & 15;
  const int hh = lane >> 4;

  const int n0 = blockIdx.x * 128;
  const int m0 = blockIdx.y * 128;
  const int b  = blockIdx.z;
  const u16* gA = A + (long long)b * aBatch;
  const u16* gB = B + (long long)b * bBatch;

  // --- staging geometry (all bounds at 16-element-chunk granularity) ---
  const int  r   = tid >> 1;             // 0..127 : A row / B-nt col
  const int  ks  = (tid & 1) * 16;       // 0 or 16
  const bool aOk = (m0 + r) < M;
  const u16* aSrc = gA + (long long)(m0 + r) * lda + ks;
  u16* aDst0 = &sA[0][r * 32 + ks];
  u16* aDst1 = &sA[1][r * 32 + ks];

  const bool btOk = (n0 + r) < N;        // transB==1 staging
  const u16* btSrc = gB + (long long)(n0 + r) * ldb + ks;
  u16* btDst0 = &sB[0][r * 32 + ks];
  u16* btDst1 = &sB[1][r * 32 + ks];

  const int  kRow = tid >> 3;            // 0..31  : transB==0 staging
  const int  nCol = (tid & 7) * 16;      // 0..112
  const bool bnOk = (n0 + nCol) < N;     // N%16==0 -> whole chunk in/out
  const u16* bnSrc = gB + (long long)kRow * ldb + n0 + nCol;

  auto stage = [&](int k0, int buf) {
    // ---- A tile ----
#if USE_ASYNC_LDS
    if (aOk) {
      u16* d = buf ? aDst1 : aDst0;
      __builtin_amdgcn_global_load_async_to_lds_b128((gvec_t)(aSrc + k0),     (lvec_t)d,       0, 0);
      __builtin_amdgcn_global_load_async_to_lds_b128((gvec_t)(aSrc + k0 + 8), (lvec_t)(d + 8), 0, 0);
    }
#else
    {
      uint4 v0 = {0, 0, 0, 0}, v1 = {0, 0, 0, 0};
      if (aOk) {
        v0 = *(const uint4*)(aSrc + k0);
        v1 = *(const uint4*)(aSrc + k0 + 8);
        __builtin_prefetch(aSrc + k0 + 32, 0, 1);
      }
      u16* d = buf ? aDst1 : aDst0;
      *(uint4*)(d)     = v0;
      *(uint4*)(d + 8) = v1;
    }
#endif
    // ---- B tile ----
    if (transB) {
#if USE_ASYNC_LDS
      if (btOk) {
        u16* d = buf ? btDst1 : btDst0;
        __builtin_amdgcn_global_load_async_to_lds_b128((gvec_t)(btSrc + k0),     (lvec_t)d,       0, 0);
        __builtin_amdgcn_global_load_async_to_lds_b128((gvec_t)(btSrc + k0 + 8), (lvec_t)(d + 8), 0, 0);
      }
#else
      uint4 v0 = {0, 0, 0, 0}, v1 = {0, 0, 0, 0};
      if (btOk) {
        v0 = *(const uint4*)(btSrc + k0);
        v1 = *(const uint4*)(btSrc + k0 + 8);
        __builtin_prefetch(btSrc + k0 + 32, 0, 1);
      }
      u16* d = buf ? btDst1 : btDst0;
      *(uint4*)(d)     = v0;
      *(uint4*)(d + 8) = v1;
#endif
    } else {
      // B[K,N]: vector-load 16 contiguous n, transpose-scatter into (n,k)
      Pack16 p;
      p.q[0] = uint4{0, 0, 0, 0};
      p.q[1] = uint4{0, 0, 0, 0};
      if (bnOk) {
        p.q[0] = *(const uint4*)(bnSrc + (long long)k0 * ldb);
        p.q[1] = *(const uint4*)(bnSrc + (long long)k0 * ldb + 8);
      }
      u16* base = &sB[buf][0];
#pragma unroll
      for (int i = 0; i < 16; ++i)
        base[(nCol + i) * 32 + kRow] = p.h[i];
    }
  };

  f32x8 acc[2][4];
#pragma unroll
  for (int i = 0; i < 2; ++i)
#pragma unroll
    for (int j = 0; j < 4; ++j)
      acc[i][j] = f32x8{0.f, 0.f, 0.f, 0.f, 0.f, 0.f, 0.f, 0.f};

  const int nk = Kd >> 5;     // K%32 == 0 guaranteed by all call sites
  stage(0, 0);
#if USE_ASYNC_LDS
  wait_async0();
#endif
  __syncthreads();

  for (int kt = 0; kt < nk; ++kt) {
    const int cur = kt & 1;
    if (kt + 1 < nk) stage((kt + 1) << 5, cur ^ 1);

    // ---- fragment loads per ISA layouts & WMMA ----
    FragAB af[2], bfr[4];
#pragma unroll
    for (int i = 0; i < 2; ++i) {
      // A 16x32 bf16: lane M=lr (half hh); K = hh*16 .. hh*16+15 (contiguous)
      const u16* pa = &sA[cur][(wm * 32 + i * 16 + lr) * 32];
      af[i].u[0] = *(const uint4*)(pa + hh * 16);
      af[i].u[1] = *(const uint4*)(pa + hh * 16 + 8);
    }
#pragma unroll
    for (int j = 0; j < 4; ++j) {
      // B 32x16 bf16: lane N=lr, K = hh*16 .. hh*16+15 (contiguous)
      const u16* pb = &sB[cur][(wn * 64 + j * 16 + lr) * 32 + hh * 16];
      bfr[j].u[0] = *(const uint4*)(pb);
      bfr[j].u[1] = *(const uint4*)(pb + 8);
    }
#pragma unroll
    for (int i = 0; i < 2; ++i)
#pragma unroll
      for (int j = 0; j < 4; ++j)
        acc[i][j] = __builtin_amdgcn_wmma_f32_16x16x32_bf16(
            false, af[i].v, false, bfr[j].v, (short)0, acc[i][j], false, false);

#if USE_ASYNC_LDS
    wait_async0();
#endif
    __syncthreads();
  }

  // ---- epilogue: C/D 16x16 f32 layout: lane N=lr, VGPR r -> M = 8*hh + r ----
  float*       gC    = C    ? C    + (long long)b * cBatch    : nullptr;
  u16*         gCb   = Cbf  ? Cbf  + (long long)b * cbfBatch  : nullptr;
  const float* gBias = bias ? bias + (long long)b * biasBatch : nullptr;
#pragma unroll
  for (int i = 0; i < 2; ++i) {
#pragma unroll
    for (int j = 0; j < 4; ++j) {
      const int n = n0 + wn * 64 + j * 16 + lr;
      if (n < N) {
#pragma unroll
        for (int rr = 0; rr < 8; ++rr) {
          const int m = m0 + wm * 32 + i * 16 + hh * 8 + rr;
          if (m < M) {
            float v = acc[i][j][rr];
            if (accumulate && gC) v += gC[(long long)m * ldc + n];
            if (gBias) v += gBias[n];
            if (act == 1) v = 0.5f * v * (1.0f + erff(v * 0.70710678118654752f));
            if (gC)  gC [(long long)m * ldc   + n] = v;
            if (gCb) gCb[(long long)m * ldcbf + n] = f2bf(v);
          }
        }
      }
    }
  }
}

// ---------------------------------------------------------------------------
// Elementwise / reduction kernels
// ---------------------------------------------------------------------------
__global__ void f32_to_bf16_k(const float* __restrict__ src, u16* __restrict__ dst,
                              long long n) {
  long long idx = (long long)blockIdx.x * blockDim.x + threadIdx.x;
  if (idx < n) dst[idx] = f2bf(src[idx]);
}

__global__ __launch_bounds__(256) void rmsnorm_k(const float* __restrict__ x,
                                                 float* __restrict__ y,
                                                 u16* __restrict__ ybf, int D) {
  const int s = blockIdx.x;
  __shared__ float red[256];
  const float* row = x + (long long)s * D;
  float ss = 0.f;
  for (int i = threadIdx.x; i < D; i += 256) { float v = row[i]; ss += v * v; }
  red[threadIdx.x] = ss; __syncthreads();
  for (int off = 128; off > 0; off >>= 1) {
    if (threadIdx.x < off) red[threadIdx.x] += red[threadIdx.x + off];
    __syncthreads();
  }
  const float scale = rsqrtf(red[0] / (float)D + 1e-6f);
  for (int i = threadIdx.x; i < D; i += 256) {
    float v = row[i] * scale;
    y  [(long long)s * D + i] = v;
    ybf[(long long)s * D + i] = f2bf(v);
  }
}

__global__ void rope_k(const float* __restrict__ in, u16* __restrict__ outbf,
                       long long total, int dim, int Sdim) {
  long long idx = (long long)blockIdx.x * blockDim.x + threadIdx.x;
  if (idx >= total) return;
  const int j    = (int)(idx % dim);
  const long long hs = idx / dim;
  const int s    = (int)(hs % Sdim);
  const int half = dim >> 1;
  const int p    = (j < half) ? j : j - half;
  const float inv = __powf(10000.f, -2.f * (float)p / (float)dim);
  const float a   = (float)s * inv;
  const float c = __cosf(a), sn = __sinf(a);
  const float rot = (j < half) ? -in[idx + half] : in[idx - half];
  outbf[idx] = f2bf(in[idx] * c + rot * sn);
}

__global__ __launch_bounds__(256) void softmax_causal_k(const float* __restrict__ scores,
                                                        u16* __restrict__ attn,
                                                        int Sdim, float scale) {
  const long long row = blockIdx.x;
  const int s = (int)(row % Sdim);
  const float* p = scores + row * Sdim;
  u16* o = attn + row * Sdim;
  __shared__ float red[256];
  float mx = -3.4e38f;
  for (int t = threadIdx.x; t < Sdim; t += 256)
    if (t <= s) mx = fmaxf(mx, p[t] * scale);
  red[threadIdx.x] = mx; __syncthreads();
  for (int off = 128; off > 0; off >>= 1) {
    if (threadIdx.x < off) red[threadIdx.x] = fmaxf(red[threadIdx.x], red[threadIdx.x + off]);
    __syncthreads();
  }
  mx = red[0]; __syncthreads();
  float sum = 0.f;
  for (int t = threadIdx.x; t < Sdim; t += 256)
    if (t <= s) sum += __expf(p[t] * scale - mx);
  red[threadIdx.x] = sum; __syncthreads();
  for (int off = 128; off > 0; off >>= 1) {
    if (threadIdx.x < off) red[threadIdx.x] += red[threadIdx.x + off];
    __syncthreads();
  }
  const float inv = 1.f / red[0];
  for (int t = threadIdx.x; t < Sdim; t += 256) {
    const float v = (t <= s) ? __expf(p[t] * scale - mx) * inv : 0.f;
    o[t] = f2bf(v);
  }
}

__global__ void add_k(const float* __restrict__ a, const float* __restrict__ b,
                      float* __restrict__ o, long long n) {
  long long idx = (long long)blockIdx.x * blockDim.x + threadIdx.x;
  if (idx < n) o[idx] = a[idx] + b[idx];
}

__global__ __launch_bounds__(256) void router_k(const float* __restrict__ y2,
                                                const float* __restrict__ Wr,
                                                const float* __restrict__ Wn,
                                                const float* __restrict__ rb,
                                                const float* __restrict__ noise,
                                                float* __restrict__ probs,
                                                int Ddim, int Edim) {
  const int s = blockIdx.x;
  __shared__ float red[256];
  __shared__ float lg[E_], nz[E_];
  const float* row = y2 + (long long)s * Ddim;
  const int e = threadIdx.x >> 5, lane = threadIdx.x & 31;
  float p1 = 0.f, p2 = 0.f;
  if (e < Edim) {
    const float* wr = Wr + (long long)e * Ddim;
    const float* wn = Wn + (long long)e * Ddim;
    for (int i = lane; i < Ddim; i += 32) {
      const float v = row[i];
      p1 += v * wr[i];
      p2 += v * wn[i];
    }
  }
  red[threadIdx.x] = p1; __syncthreads();
  for (int off = 16; off > 0; off >>= 1) {
    if (lane < off) red[threadIdx.x] += red[threadIdx.x + off];
    __syncthreads();
  }
  if (lane == 0 && e < Edim) lg[e] = red[threadIdx.x];
  __syncthreads();
  red[threadIdx.x] = p2; __syncthreads();
  for (int off = 16; off > 0; off >>= 1) {
    if (lane < off) red[threadIdx.x] += red[threadIdx.x + off];
    __syncthreads();
  }
  if (lane == 0 && e < Edim) nz[e] = red[threadIdx.x];
  __syncthreads();
  if (threadIdx.x == 0) {
    float v[E_];
    for (int i = 0; i < Edim; ++i) {
      const float z  = nz[i];
      const float sp = (z > 20.f) ? z : log1pf(__expf(z));
      v[i] = noise[(long long)s * Edim + i] * sp + lg[i] + rb[i];
    }
    int i0 = 0;
    for (int i = 1; i < Edim; ++i) if (v[i] > v[i0]) i0 = i;
    int i1 = (i0 == 0) ? 1 : 0;
    for (int i = 0; i < Edim; ++i) { if (i == i0) continue; if (v[i] > v[i1]) i1 = i; }
    const float m  = fmaxf(v[i0], v[i1]);
    const float e0 = __expf(v[i0] - m), e1 = __expf(v[i1] - m);
    const float den = e0 + e1;
    for (int i = 0; i < Edim; ++i) probs[(long long)s * Edim + i] = 0.f;
    probs[(long long)s * Edim + i0] = e0 / den;
    probs[(long long)s * Edim + i1] = e1 / den;
  }
}

__global__ void combine_k(const float* __restrict__ xmid, const float* __restrict__ probs,
                          const float* __restrict__ EO, float* __restrict__ out,
                          int Sdim, int Ddim, int Edim) {
  const long long n = (long long)Sdim * Ddim;
  long long idx = (long long)blockIdx.x * blockDim.x + threadIdx.x;
  if (idx >= n) return;
  const int s = (int)(idx / Ddim);
  float acc = xmid[idx];
  for (int e = 0; e < Edim; ++e)
    acc += probs[(long long)s * Edim + e] * EO[(long long)e * n + idx];
  out[idx] = acc;
}

// ---------------------------------------------------------------------------
// Host orchestration
// ---------------------------------------------------------------------------
extern "C" void kernel_launch(void* const* d_in, const int* in_sizes, int n_in,
                              void* d_out, int out_size, void* d_ws, size_t ws_size,
                              hipStream_t stream) {
  (void)in_sizes; (void)n_in; (void)out_size; (void)ws_size;
  const float* x      = (const float*)d_in[0];
  const float* noise  = (const float*)d_in[1];
  const float* Wdkv   = (const float*)d_in[2];
  const float* Wq     = (const float*)d_in[3];
  const float* Wuv    = (const float*)d_in[4];
  const float* Wuk    = (const float*)d_in[5];
  const float* Wkr    = (const float*)d_in[6];
  const float* Wqr    = (const float*)d_in[7];
  const float* bqr    = (const float*)d_in[8];
  const float* Wo     = (const float*)d_in[9];
  const float* Wr     = (const float*)d_in[10];
  const float* Wn     = (const float*)d_in[11];
  const float* rbias  = (const float*)d_in[12];
  const float* W1     = (const float*)d_in[13];
  const float* b1     = (const float*)d_in[14];
  const float* W2     = (const float*)d_in[15];
  const float* b2     = (const float*)d_in[16];
  float* out = (float*)d_out;

  size_t cur = 0;
  auto alloc = [&](size_t elems, size_t esz) -> void* {
    cur = (cur + 255) & ~(size_t)255;
    void* p = (char*)d_ws + cur;
    cur += elems * esz;
    return p;
  };

  const long long SD  = (long long)S_ * D_;
  const long long HSL = (long long)H_ * S_ * L_;
  const long long HSS = (long long)H_ * S_ * S_;

  float* y1      = (float*)alloc(SD, 4);
  u16*   y1b     = (u16*)  alloc(SD, 2);
  u16*   Wdkv_b  = (u16*)  alloc(H_ * L_ * HD_, 2);
  u16*   Wq_b    = (u16*)  alloc(H_ * HD_ * HD_, 2);
  u16*   Wuv_b   = (u16*)  alloc(H_ * L_ * HD_, 2);
  u16*   Wuk_b   = (u16*)  alloc(H_ * L_ * HD_, 2);
  u16*   Wkr_b   = (u16*)  alloc(H_ * L_ * HD_, 2);
  u16*   Wqr_b   = (u16*)  alloc(H_ * HD_ * L_, 2);
  u16*   Wo_b    = (u16*)  alloc(H_ * HD_ * HD_, 2);
  u16*   W1_b    = (u16*)  alloc((long long)E_ * D_ * F_, 2);
  u16*   W2_b    = (u16*)  alloc((long long)E_ * F_ * D_, 2);
  u16*   ckv_b   = (u16*)  alloc(HSL, 2);
  float* kf      = (float*)alloc(HSL, 4);
  u16*   k_b     = (u16*)  alloc(HSL, 2);
  float* qf      = (float*)alloc((long long)H_ * S_ * HD_, 4);
  u16*   q_b     = (u16*)  alloc((long long)H_ * S_ * HD_, 2);
  u16*   abs_b   = (u16*)  alloc(H_ * HD_ * L_, 2);
  u16*   t_b     = (u16*)  alloc(HSL, 2);
  float* scores  = (float*)alloc(HSS, 4);
  u16*   attn_b  = (u16*)  alloc(HSS, 2);
  u16*   v_b     = (u16*)  alloc((long long)H_ * S_ * HD_, 2);
  u16*   o_b     = (u16*)  alloc((long long)H_ * S_ * HD_, 2);
  float* attnout = (float*)alloc(SD, 4);
  float* xmid    = (float*)alloc(SD, 4);
  float* y2      = (float*)alloc(SD, 4);
  u16*   y2b     = (u16*)  alloc(SD, 2);
  float* probs   = (float*)alloc((long long)S_ * E_, 4);
  u16*   Hbuf    = (u16*)  alloc((long long)E_ * S_ * F_, 2);
  float* EO      = (float*)alloc((long long)E_ * S_ * D_, 4);

  auto cvt = [&](const float* src, u16* dst, long long n) {
    const int blk = 256;
    f32_to_bf16_k<<<(unsigned)((n + blk - 1) / blk), blk, 0, stream>>>(src, dst, n);
  };
  auto gemm = [&](const u16* A, long long aB, int lda,
                  const u16* B, long long bB, int ldb, int tB,
                  float* C, long long cB, int ldc,
                  u16* Cbf, long long cbfB, int ldcbf,
                  const float* bias, long long biasB,
                  int M, int N, int Kd, int accum, int act, int batch) {
    dim3 g((N + 127) / 128, (M + 127) / 128, batch);
    wmma_gemm<<<g, 256, 0, stream>>>(A, aB, lda, B, bB, ldb, tB,
                                     C, cB, ldc, Cbf, cbfB, ldcbf,
                                     bias, biasB, M, N, Kd, accum, act);
  };

  // ---- weight conversions to bf16 ----
  cvt(Wdkv, Wdkv_b, H_ * L_ * HD_);
  cvt(Wq,   Wq_b,   H_ * HD_ * HD_);
  cvt(Wuv,  Wuv_b,  H_ * L_ * HD_);
  cvt(Wuk,  Wuk_b,  H_ * L_ * HD_);
  cvt(Wkr,  Wkr_b,  H_ * L_ * HD_);
  cvt(Wqr,  Wqr_b,  H_ * HD_ * L_);
  cvt(Wo,   Wo_b,   H_ * HD_ * HD_);
  cvt(W1,   W1_b,   (long long)E_ * D_ * F_);
  cvt(W2,   W2_b,   (long long)E_ * F_ * D_);

  // ---- attention ----
  rmsnorm_k<<<S_, 256, 0, stream>>>(x, y1, y1b, D_);

  // ckv[h] = xs_h @ Wdkv[h]^T        (bf16 only)
  gemm(y1b, 64, D_, Wdkv_b, (long long)L_ * HD_, HD_, 1,
       nullptr, 0, 0, ckv_b, (long long)S_ * L_, L_,
       nullptr, 0, S_, L_, HD_, 0, 0, H_);
  // k[h] = xs_h @ Wk_rope[h]^T       (f32 -> rope)
  gemm(y1b, 64, D_, Wkr_b, (long long)L_ * HD_, HD_, 1,
       kf, (long long)S_ * L_, L_, nullptr, 0, 0,
       nullptr, 0, S_, L_, HD_, 0, 0, H_);
  rope_k<<<(unsigned)((HSL + 255) / 256), 256, 0, stream>>>(kf, k_b, HSL, L_, S_);
  // q[h] = ckv[h] @ Wq_rope[h]^T + bq_rope[h]   (f32 -> rope)
  gemm(ckv_b, (long long)S_ * L_, L_, Wqr_b, (long long)HD_ * L_, L_, 1,
       qf, (long long)S_ * HD_, HD_, nullptr, 0, 0,
       bqr, HD_, S_, HD_, L_, 0, 0, H_);
  rope_k<<<(unsigned)(((long long)H_ * S_ * HD_ + 255) / 256), 256, 0, stream>>>(
      qf, q_b, (long long)H_ * S_ * HD_, HD_, S_);
  // absorbed[h] = Wq[h] @ Wuk[h]^T   (bf16 only)
  gemm(Wq_b, (long long)HD_ * HD_, HD_, Wuk_b, (long long)L_ * HD_, HD_, 1,
       nullptr, 0, 0, abs_b, (long long)HD_ * L_, L_,
       nullptr, 0, HD_, L_, HD_, 0, 0, H_);
  // t[h] = xs_h @ absorbed[h]        (bf16 only)
  gemm(y1b, 64, D_, abs_b, (long long)HD_ * L_, L_, 0,
       nullptr, 0, 0, t_b, (long long)S_ * L_, L_,
       nullptr, 0, S_, L_, HD_, 0, 0, H_);
  // s1 = t @ ckv^T
  gemm(t_b, (long long)S_ * L_, L_, ckv_b, (long long)S_ * L_, L_, 1,
       scores, (long long)S_ * S_, S_, nullptr, 0, 0,
       nullptr, 0, S_, S_, L_, 0, 0, H_);
  // s2 = q @ k^T  (accumulate into scores)
  gemm(q_b, (long long)S_ * HD_, HD_, k_b, (long long)S_ * L_, L_, 1,
       scores, (long long)S_ * S_, S_, nullptr, 0, 0,
       nullptr, 0, S_, S_, HD_, 1, 0, H_);
  // softmax((s1+s2)/sqrt(2*HD)) with causal mask -> attn bf16
  softmax_causal_k<<<H_ * S_, 256, 0, stream>>>(scores, attn_b, S_,
                                                1.0f / sqrtf(2.0f * HD_));
  // v[h] = ckv[h] @ Wuv[h]           (bf16 only; NN: Wuv[h] is [L,HD])
  gemm(ckv_b, (long long)S_ * L_, L_, Wuv_b, (long long)L_ * HD_, HD_, 0,
       nullptr, 0, 0, v_b, (long long)S_ * HD_, HD_,
       nullptr, 0, S_, HD_, L_, 0, 0, H_);
  // o[h] = attn[h] @ v[h]            (bf16 only)
  gemm(attn_b, (long long)S_ * S_, S_, v_b, (long long)S_ * HD_, HD_, 0,
       nullptr, 0, 0, o_b, (long long)S_ * HD_, HD_,
       nullptr, 0, S_, HD_, S_, 0, 0, H_);
  // attn_out[:, h*64:(h+1)*64] = o[h] @ Wo[h]^T
  gemm(o_b, (long long)S_ * HD_, HD_, Wo_b, (long long)HD_ * HD_, HD_, 1,
       attnout, 64, D_, nullptr, 0, 0,
       nullptr, 0, S_, HD_, HD_, 0, 0, H_);
  // residual
  add_k<<<(unsigned)((SD + 255) / 256), 256, 0, stream>>>(x, attnout, xmid, SD);

  // ---- MoE ----
  rmsnorm_k<<<S_, 256, 0, stream>>>(xmid, y2, y2b, D_);
  router_k<<<S_, 256, 0, stream>>>(y2, Wr, Wn, rbias, noise, probs, D_, E_);
  // fc1 (all experts batched): H[e] = gelu(y2 @ W1[e] + b1[e])  (bf16 only)
  gemm(y2b, 0, D_, W1_b, (long long)D_ * F_, F_, 0,
       nullptr, 0, 0, Hbuf, (long long)S_ * F_, F_,
       b1, F_, S_, F_, D_, 0, 1, E_);
  // fc2: EO[e] = H[e] @ W2[e] + b2[e]  (f32)
  gemm(Hbuf, (long long)S_ * F_, F_, W2_b, (long long)F_ * D_, D_, 0,
       EO, (long long)S_ * D_, D_, nullptr, 0, 0,
       b2, D_, S_, D_, F_, 0, 0, E_);
  // out = xmid + sum_e probs[:,e] * EO[e]
  combine_k<<<(unsigned)((SD + 255) / 256), 256, 0, stream>>>(xmid, probs, EO, out,
                                                              S_, D_, E_);
}
